// GATFraudDetector_32925219291865
// MI455X (gfx1250) — matrix operand order repaired and checked by reading, not compiled, not run on hardware
//
#include <hip/hip_runtime.h>
#include <hip/hip_bf16.h>

// ---------------------------------------------------------------------------
// GAT fraud detector for MI455X (gfx1250, wave32, WMMA).
// - Dense projections: bf16 v_wmma_f32_16x16x32_bf16, fp32 accumulate.
// - Edge softmax: edge-parallel 3-pass with tiny N*H atomics.
// - Aggregation: destination-CSR (built once), wave-per-node, atomic-free,
//   register-resident 256-wide accumulator. Feature matrix (51MB) is
//   L2-resident on the 192MB L2, so gathers mostly hit L2.
// ---------------------------------------------------------------------------

typedef __attribute__((ext_vector_type(16))) __bf16 v16bf;
typedef __attribute__((ext_vector_type(8)))  float  v8f;

#define NEG_SLOPE 0.2f
#define BN_EPS    1e-5f

// ---------------------------------------------------------------- utilities

__global__ void fill_f32(float* __restrict__ p, float v, long long n) {
    long long i = (long long)blockIdx.x * blockDim.x + threadIdx.x;
    if (i < n) p[i] = v;
}

__global__ void fill_i32(int* __restrict__ p, int v, long long n) {
    long long i = (long long)blockIdx.x * blockDim.x + threadIdx.x;
    if (i < n) p[i] = v;
}

__global__ void copy_i32(const int* __restrict__ in, int* __restrict__ out, long long n) {
    long long i = (long long)blockIdx.x * blockDim.x + threadIdx.x;
    if (i < n) out[i] = in[i];
}

__global__ void cvt_f32_bf16(const float* __restrict__ in, __bf16* __restrict__ out,
                             long long n) {
    long long i = (long long)blockIdx.x * blockDim.x + threadIdx.x;
    if (i < n) out[i] = (__bf16)in[i];
}

// W [K, Nc] f32 row-major  ->  Wt [Nc, K] bf16 row-major (K-contiguous per column)
__global__ void cvt_w_transpose(const float* __restrict__ W, __bf16* __restrict__ Wt,
                                int K, int Nc) {
    int i = blockIdx.x * blockDim.x + threadIdx.x;
    if (i >= K * Nc) return;
    int k = i / Nc, n = i % Nc;
    Wt[(size_t)n * K + k] = (__bf16)W[i];
}

__device__ inline void atomicMaxF(float* addr, float val) {
    unsigned int* u = (unsigned int*)addr;
    unsigned int old = *u;
    while (__uint_as_float(old) < val) {
        unsigned int assumed = old;
        old = atomicCAS(u, assumed, __float_as_uint(val));
        if (old == assumed) break;
    }
}

// ------------------------------------------------------------- WMMA GEMM
// C[M,Nc] f32 = A[M,K] bf16  @  Wt[Nc,K]^T  (one wave per 16x16 tile)

#define GEMM_WAVES 4  // waves per block, each takes a different M tile

__global__ __launch_bounds__(GEMM_WAVES * 32)
void wmma_gemm_bf16(const __bf16* __restrict__ A, const __bf16* __restrict__ Bt,
                    float* __restrict__ C, int M, int K, int Nc) {
    const int lane = threadIdx.x & 31;
    const int wave = threadIdx.x >> 5;
    const int mt   = blockIdx.x * GEMM_WAVES + wave;   // M tile
    const int m0   = mt * 16;
    if (m0 >= M) return;                               // wave-uniform exit
    const int n0   = blockIdx.y * 16;                  // N tile
    const int half = lane >> 4;                        // 0: lanes 0-15, 1: 16-31
    const int arow = min(m0 + (lane & 15), M - 1);     // clamp (padded rows dup)
    const int ncol = n0 + (lane & 15);

    const __bf16* arow_p = A  + (size_t)arow * K;
    const __bf16* bcol_p = Bt + (size_t)ncol * K;

    union Frag { v16bf v; unsigned int u[8]; };
    v8f acc = {};

    for (int k0 = 0; k0 < K; k0 += 32) {
        Frag a, b;
        // A fragment: 16-bit A 16x32 layout — VGPR v: K = (v>=4?16:0) + half*8 + 2*(v&3)
#pragma unroll
        for (int v = 0; v < 8; ++v) {
            int kk = k0 + ((v & 4) << 2) + half * 8 + ((v & 3) << 1);
            a.u[v] = *(const unsigned int*)(arow_p + kk);
        }
        // B fragment: 16-bit B 32x16 layout — VGPR v: K = (v>=4?16:0) + half*8 + 2*(v&3), column = ncol
#pragma unroll
        for (int v = 0; v < 8; ++v) {
            int kk = k0 + ((v & 4) << 2) + half * 8 + ((v & 3) << 1);
            b.u[v] = *(const unsigned int*)(bcol_p + kk);
        }
        acc = __builtin_amdgcn_wmma_f32_16x16x32_bf16(
            false, a.v, false, b.v, (short)0, acc, false, false);
    }

    // C/D layout: VGPR r -> M = m0 + half*8 + r, column ncol
#pragma unroll
    for (int r = 0; r < 8; ++r) {
        int mr = m0 + half * 8 + r;
        if (mr < M) C[(size_t)mr * Nc + ncol] = acc[r];
    }
}

// ------------------------------------------------- attention coefficients
// s[n,h] = sum_c feat[n,h*64+c]*asrc[h*64+c] ; d likewise (Hd = 64 fixed)

__global__ void attn_sd(const float* __restrict__ feat, const float* __restrict__ asrc,
                        const float* __restrict__ adst, float* __restrict__ s,
                        float* __restrict__ d, int Nn, int H) {
    int i = blockIdx.x * blockDim.x + threadIdx.x;
    if (i >= Nn * H) return;
    int h = i % H, n = i / H;
    const float* row = feat + (size_t)n * H * 64 + h * 64;
    const float* as  = asrc + h * 64;
    const float* ad  = adst + h * 64;
    float ss = 0.f, dd = 0.f;
#pragma unroll 8
    for (int c = 0; c < 64; ++c) { float v = row[c]; ss += v * as[c]; dd += v * ad[c]; }
    s[i] = ss; d[i] = dd;
}

// --------------------------------------------------- CSR build (once per call)
// Edges 0..E-1 from edge_index (int64), E..E+N-1 are self-loops.

__global__ void count_deg(const long long* __restrict__ ei, int* __restrict__ deg,
                          int E, int Nn) {
    long long i = (long long)blockIdx.x * blockDim.x + threadIdx.x;
    if (i >= (long long)E + Nn) return;
    int dst = (i < E) ? (int)ei[E + i] : (int)(i - E);
    atomicAdd(&deg[dst], 1);
}

#define SCAN_B 256

__global__ __launch_bounds__(SCAN_B)
void scan_block(const int* __restrict__ in, int* __restrict__ out,
                int* __restrict__ partials, int n) {
    __shared__ int sh[SCAN_B];
    int gid = blockIdx.x * SCAN_B + threadIdx.x;
    int v = gid < n ? in[gid] : 0;
    sh[threadIdx.x] = v;
    __syncthreads();
    for (int off = 1; off < SCAN_B; off <<= 1) {       // Hillis-Steele inclusive
        int t = threadIdx.x >= off ? sh[threadIdx.x - off] : 0;
        __syncthreads();
        sh[threadIdx.x] += t;
        __syncthreads();
    }
    if (gid < n) out[gid] = sh[threadIdx.x] - v;       // exclusive
    if (threadIdx.x == SCAN_B - 1) partials[blockIdx.x] = sh[threadIdx.x];
}

__global__ __launch_bounds__(1024)
void scan_partials(int* __restrict__ partials, int nb) {  // nb <= 1024
    __shared__ int sh[1024];
    int t = threadIdx.x;
    int v = t < nb ? partials[t] : 0;
    sh[t] = v;
    __syncthreads();
    for (int off = 1; off < 1024; off <<= 1) {
        int tv = t >= off ? sh[t - off] : 0;
        __syncthreads();
        sh[t] += tv;
        __syncthreads();
    }
    if (t < nb) partials[t] = sh[t] - v;               // exclusive
}

__global__ __launch_bounds__(SCAN_B)
void scan_add(int* __restrict__ out, const int* __restrict__ partials, int n) {
    int gid = blockIdx.x * SCAN_B + threadIdx.x;
    if (gid < n) out[gid] += partials[blockIdx.x];
}

__global__ void csr_fill(const long long* __restrict__ ei, int* __restrict__ cursor,
                         int* __restrict__ csr_src, int* __restrict__ csr_eid,
                         int E, int Nn) {
    long long i = (long long)blockIdx.x * blockDim.x + threadIdx.x;
    if (i >= (long long)E + Nn) return;
    int src, dst;
    if (i < E) { src = (int)ei[i]; dst = (int)ei[E + i]; }
    else       { src = dst = (int)(i - E); }
    int pos = atomicAdd(&cursor[dst], 1);
    csr_src[pos] = src;
    csr_eid[pos] = (int)i;
}

// ----------------------------------------------------------- edge softmax

__global__ void edge_pass1(const long long* __restrict__ ei, const float* __restrict__ s,
                           const float* __restrict__ d, float* __restrict__ m,
                           float* __restrict__ e, int E, int Nn, int H) {
    long long tot = (long long)(E + Nn) * H;
    long long i = (long long)blockIdx.x * blockDim.x + threadIdx.x;
    if (i >= tot) return;
    int h = (int)(i % H);
    long long idx = i / H;
    int src, dst;
    if (idx < E) { src = (int)ei[idx]; dst = (int)ei[E + idx]; }
    else         { src = dst = (int)(idx - E); }
    float v = s[(size_t)src * H + h] + d[(size_t)dst * H + h];
    v = v > 0.f ? v : NEG_SLOPE * v;                    // leaky relu
    e[i] = v;
    atomicMaxF(&m[(size_t)dst * H + h], v);
}

__global__ void edge_pass2(const long long* __restrict__ ei, const float* __restrict__ m,
                           float* __restrict__ den, float* __restrict__ e,
                           int E, int Nn, int H) {
    long long tot = (long long)(E + Nn) * H;
    long long i = (long long)blockIdx.x * blockDim.x + threadIdx.x;
    if (i >= tot) return;
    int h = (int)(i % H);
    long long idx = i / H;
    int dst = (idx < E) ? (int)ei[E + idx] : (int)(idx - E);
    float v = __expf(e[i] - m[(size_t)dst * H + h]);
    e[i] = v;
    atomicAdd(&den[(size_t)dst * H + h], v);
}

// ----------------------------------------- atomic-free CSR aggregation
// One wave32 per destination node; NF = Fout/32 accumulator VGPRs per lane.
// Head index for feature f = lane + 32*j is lane-invariant: h = j>>1 (Hd=64).

template <int NF, int H>
__global__ __launch_bounds__(256)
void aggregate_csr(const int* __restrict__ offs, const int* __restrict__ deg,
                   const int* __restrict__ csr_src, const int* __restrict__ csr_eid,
                   const float* __restrict__ eb, const float* __restrict__ den,
                   const float* __restrict__ feat, const float* __restrict__ bias,
                   float* __restrict__ out, int Nn) {
    const int Fout = NF * 32;
    const int lane = threadIdx.x & 31;
    const int n = blockIdx.x * (blockDim.x >> 5) + (threadIdx.x >> 5);
    if (n >= Nn) return;                               // wave-uniform exit

    float invd[H];
#pragma unroll
    for (int h = 0; h < H; ++h)
        invd[h] = 1.f / (den[(size_t)n * H + h] + 1e-16f);

    float acc[NF];
#pragma unroll
    for (int j = 0; j < NF; ++j) acc[j] = 0.f;

    const int start = offs[n];
    const int cnt   = deg[n];
    for (int p = start; p < start + cnt; ++p) {
        const int srcn = csr_src[p];
        const int eidx = csr_eid[p];
        const float* frow = feat + (size_t)srcn * Fout;
        __builtin_prefetch(frow, 0, 1);                // global_prefetch_b8
        float al[H];
#pragma unroll
        for (int h = 0; h < H; ++h)
            al[h] = eb[(size_t)eidx * H + h] * invd[h];
#pragma unroll
        for (int j = 0; j < NF; ++j)
            acc[j] = fmaf(al[(j >> 1) % H], frow[lane + (j << 5)], acc[j]);
    }
#pragma unroll
    for (int j = 0; j < NF; ++j) {
        int f = lane + (j << 5);
        out[(size_t)n * Fout + f] = acc[j] + bias[f];
    }
}

// -------------------------------------------------------------- batchnorm

__global__ __launch_bounds__(256)
void bn_stats(const float* __restrict__ x, float* __restrict__ mean,
              float* __restrict__ var, int Nn, int C) {
    int c = blockIdx.x;
    float s = 0.f, s2 = 0.f;
    for (int n = threadIdx.x; n < Nn; n += blockDim.x) {
        float v = x[(size_t)n * C + c];
        s += v; s2 += v * v;
    }
    __shared__ float sh[256], sh2[256];
    sh[threadIdx.x] = s; sh2[threadIdx.x] = s2;
    __syncthreads();
    for (int off = 128; off > 0; off >>= 1) {
        if (threadIdx.x < off) {
            sh[threadIdx.x]  += sh[threadIdx.x + off];
            sh2[threadIdx.x] += sh2[threadIdx.x + off];
        }
        __syncthreads();
    }
    if (threadIdx.x == 0) {
        float mu = sh[0] / Nn;
        mean[c] = mu;
        var[c]  = sh2[0] / Nn - mu * mu;               // biased var (jnp default)
    }
}

// y = relu(gamma*(x-mu)*rsqrt(var+eps)+beta), write f32 + bf16 (next-layer GEMM A)
__global__ void bn_relu_cvt(const float* __restrict__ x, const float* __restrict__ mean,
                            const float* __restrict__ var, const float* __restrict__ g,
                            const float* __restrict__ be, float* __restrict__ yf,
                            __bf16* __restrict__ yb, int Nn, int C) {
    long long i = (long long)blockIdx.x * blockDim.x + threadIdx.x;
    if (i >= (long long)Nn * C) return;
    int c = (int)(i % C);
    float v = (x[i] - mean[c]) * rsqrtf(var[c] + BN_EPS) * g[c] + be[c];
    v = v > 0.f ? v : 0.f;
    yf[i] = v;
    yb[i] = (__bf16)v;
}

// ------------------------------------------------------- pooling + classifier

__global__ __launch_bounds__(256)
void pool_mean_max(const float* __restrict__ x, float* __restrict__ pooled,
                   int Nn, int C) {
    int c = blockIdx.x;
    float s = 0.f, mx = -1e30f;
    for (int n = threadIdx.x; n < Nn; n += blockDim.x) {
        float v = x[(size_t)n * C + c];
        s += v; mx = fmaxf(mx, v);
    }
    __shared__ float sh[256], shm[256];
    sh[threadIdx.x] = s; shm[threadIdx.x] = mx;
    __syncthreads();
    for (int off = 128; off > 0; off >>= 1) {
        if (threadIdx.x < off) {
            sh[threadIdx.x]  += sh[threadIdx.x + off];
            shm[threadIdx.x]  = fmaxf(shm[threadIdx.x], shm[threadIdx.x + off]);
        }
        __syncthreads();
    }
    if (threadIdx.x == 0) { pooled[c] = sh[0] / Nn; pooled[C + c] = shm[0]; }
}

__global__ __launch_bounds__(128)
void classifier(const float* __restrict__ pooled, const float* __restrict__ Wc1,
                const float* __restrict__ bc1, const float* __restrict__ Wc2,
                const float* __restrict__ bc2, float* __restrict__ out) {
    __shared__ float pl[128], z[64];
    int t = threadIdx.x;
    pl[t] = pooled[t];
    __syncthreads();
    if (t < 64) {
        float a = bc1[t];
        for (int i = 0; i < 128; ++i) a += pl[i] * Wc1[i * 64 + t];
        z[t] = a > 0.f ? a : 0.f;
    }
    __syncthreads();
    if (t < 2) {
        float a = bc2[t];
        for (int j = 0; j < 64; ++j) a += z[j] * Wc2[j * 2 + t];
        out[t] = a;
    }
}

// ---------------------------------------------------------------- driver

static inline unsigned int ceil_div_ll(long long a, long long b) {
    return (unsigned int)((a + b - 1) / b);
}

extern "C" void kernel_launch(void* const* d_in, const int* in_sizes, int n_in,
                              void* d_out, int out_size, void* d_ws, size_t ws_size,
                              hipStream_t stream) {
    const int Fin  = 128, WD = 256, Hd = 64;
    const int N    = in_sizes[0] / Fin;
    const int E    = in_sizes[1] / 2;
    const long long Et = (long long)E + N;

    const float*     x  = (const float*)d_in[0];
    const long long* ei = (const long long*)d_in[1];
    const float* P[3][6];   // W, b, asrc, adst, gamma, beta
    for (int l = 0; l < 3; ++l)
        for (int j = 0; j < 6; ++j) P[l][j] = (const float*)d_in[2 + l * 6 + j];
    const float* Wc1 = (const float*)d_in[20];
    const float* bc1 = (const float*)d_in[21];
    const float* Wc2 = (const float*)d_in[22];
    const float* bc2 = (const float*)d_in[23];
    float* out = (float*)d_out;

    // ---- workspace carve-out (256B aligned) ----
    char* p = (char*)d_ws;
    auto carve = [&](size_t bytes) {
        void* r = (void*)p;
        p += (bytes + 255) & ~(size_t)255;
        return r;
    };
    __bf16* Abf  = (__bf16*)carve((size_t)N * WD * 2);   // bf16 layer input
    __bf16* Wt   = (__bf16*)carve((size_t)WD * WD * 2);  // bf16 transposed weight
    float*  Hb   = (float*) carve((size_t)N * WD * 4);   // projected features h
    float*  Ob   = (float*) carve((size_t)N * WD * 4);   // aggregated output
    float*  sb   = (float*) carve((size_t)N * 4 * 4);
    float*  db   = (float*) carve((size_t)N * 4 * 4);
    float*  mb   = (float*) carve((size_t)N * 4 * 4);
    float*  den  = (float*) carve((size_t)N * 4 * 4);
    float*  eb   = (float*) carve((size_t)Et * 4 * 4);
    int*    deg  = (int*)   carve((size_t)N * 4);
    int*    offs = (int*)   carve((size_t)N * 4);
    int*    cur  = (int*)   carve((size_t)N * 4);
    int*    prt  = (int*)   carve(1024 * 4);
    int*    csrS = (int*)   carve((size_t)Et * 4);
    int*    csrE = (int*)   carve((size_t)Et * 4);
    float*  mean = (float*) carve(WD * 4);
    float*  var  = (float*) carve(WD * 4);
    float*  pooled = (float*)carve(2 * Hd * 4);
    (void)ws_size; (void)n_in; (void)out_size;

    // ---- build destination-CSR once (graph identical for all 3 layers) ----
    fill_i32<<<ceil_div_ll(N, 256), 256, 0, stream>>>(deg, 0, N);
    count_deg<<<ceil_div_ll(Et, 256), 256, 0, stream>>>(ei, deg, E, N);
    const int nb = (int)ceil_div_ll(N, SCAN_B);          // <= 1024 for N <= 262144
    scan_block<<<nb, SCAN_B, 0, stream>>>(deg, offs, prt, N);
    scan_partials<<<1, 1024, 0, stream>>>(prt, nb);
    scan_add<<<nb, SCAN_B, 0, stream>>>(offs, prt, N);
    copy_i32<<<ceil_div_ll(N, 256), 256, 0, stream>>>(offs, cur, N);
    csr_fill<<<ceil_div_ll(Et, 256), 256, 0, stream>>>(ei, cur, csrS, csrE, E, N);

    auto run_gat = [&](const __bf16* in_bf, int K, int H,
                       const float* W, const float* b,
                       const float* asrc, const float* adst) {
        const int Fout = H * Hd;
        cvt_w_transpose<<<ceil_div_ll((long long)K * Fout, 256), 256, 0, stream>>>(
            W, Wt, K, Fout);
        dim3 gg(ceil_div_ll(((long long)N + 15) / 16, GEMM_WAVES), Fout / 16);
        wmma_gemm_bf16<<<gg, GEMM_WAVES * 32, 0, stream>>>(in_bf, Wt, Hb, N, K, Fout);
        attn_sd<<<ceil_div_ll((long long)N * H, 256), 256, 0, stream>>>(
            Hb, asrc, adst, sb, db, N, H);
        fill_f32<<<ceil_div_ll((long long)N * H, 256), 256, 0, stream>>>(
            mb, -1e30f, (long long)N * H);
        fill_f32<<<ceil_div_ll((long long)N * H, 256), 256, 0, stream>>>(
            den, 0.f, (long long)N * H);
        edge_pass1<<<ceil_div_ll(Et * H, 256), 256, 0, stream>>>(ei, sb, db, mb, eb, E, N, H);
        edge_pass2<<<ceil_div_ll(Et * H, 256), 256, 0, stream>>>(ei, mb, den, eb, E, N, H);
        // atomic-free aggregation, bias folded in
        if (H == 4)
            aggregate_csr<8, 4><<<ceil_div_ll(N, 8), 256, 0, stream>>>(
                offs, deg, csrS, csrE, eb, den, Hb, b, Ob, N);
        else
            aggregate_csr<2, 1><<<ceil_div_ll(N, 8), 256, 0, stream>>>(
                offs, deg, csrS, csrE, eb, den, Hb, b, Ob, N);
        bn_stats<<<Fout, 256, 0, stream>>>(Ob, mean, var, N, Fout);
    };

    // layer 0: 128 -> 4x64 concat
    cvt_f32_bf16<<<ceil_div_ll((long long)N * Fin, 256), 256, 0, stream>>>(
        x, Abf, (long long)N * Fin);
    run_gat(Abf, Fin, 4, P[0][0], P[0][1], P[0][2], P[0][3]);
    bn_relu_cvt<<<ceil_div_ll((long long)N * WD, 256), 256, 0, stream>>>(
        Ob, mean, var, P[0][4], P[0][5], Hb, Abf, N, WD);

    // layer 1: 256 -> 4x64 concat
    run_gat(Abf, WD, 4, P[1][0], P[1][1], P[1][2], P[1][3]);
    bn_relu_cvt<<<ceil_div_ll((long long)N * WD, 256), 256, 0, stream>>>(
        Ob, mean, var, P[1][4], P[1][5], Hb, Abf, N, WD);

    // layer 2: 256 -> 1x64 (mean over single head == identity)
    run_gat(Abf, WD, 1, P[2][0], P[2][1], P[2][2], P[2][3]);
    bn_relu_cvt<<<ceil_div_ll((long long)N * Hd, 256), 256, 0, stream>>>(
        Ob, mean, var, P[2][4], P[2][5], Hb, Abf, N, Hd);

    // readout: [mean | max] pooling + 2-layer MLP
    pool_mean_max<<<Hd, 256, 0, stream>>>(Hb, pooled, N, Hd);
    classifier<<<1, 128, 0, stream>>>(pooled, Wc1, bc1, Wc2, bc2, out);
}
